// MultiHeadAttentionWindow_42073499631674
// MI455X (gfx1250) — matrix operand order repaired and checked by reading, not compiled, not run on hardware
//
#include <hip/hip_runtime.h>
#include <hip/hip_bf16.h>

typedef __attribute__((ext_vector_type(16))) _Float16 v16h;
typedef __attribute__((ext_vector_type(8)))  float    v8f;
typedef __attribute__((ext_vector_type(4)))  _Float16 v4h;
typedef __attribute__((ext_vector_type(2)))  _Float16 v2h;
typedef __attribute__((ext_vector_type(4)))  int      v4i;

typedef __attribute__((address_space(1))) v4i gas_v4i;
typedef __attribute__((address_space(3))) v4i las_v4i;

#if defined(__has_builtin)
#if __has_builtin(__builtin_amdgcn_global_load_async_to_lds_b128)
#define HAVE_ASYNC_LDS 1
#endif
#endif

__device__ __forceinline__ void async_copy16(const void* g, void* l) {
#ifdef HAVE_ASYNC_LDS
    __builtin_amdgcn_global_load_async_to_lds_b128((gas_v4i*)g, (las_v4i*)l,
                                                   0, 0);
#endif
}

__device__ __forceinline__ void wait_async0() {
#if defined(__has_builtin) && __has_builtin(__builtin_amdgcn_s_wait_asynccnt)
    __builtin_amdgcn_s_wait_asynccnt(0);
#else
    asm volatile("s_wait_asynccnt 0x0" ::: "memory");
#endif
}

#define B_    4
#define K_    8400
#define DM    512
#define H_    8
#define WIN_  168
#define PAD_  42
#define STEP_ 84
#define KP_   (K_ + 2 * PAD_)   // 8484
#define NWIN  (K_ / STEP_)      // 100
#define HB_   (H_ * B_)         // 32
#define MTOT  (B_ * KP_)        // 33936
#define NQKV  (3 * DM)          // 1536
#define MOUT  (B_ * K_)         // 33600

// A-matrix 16x32 f16 fragment K-index for element e (0..15), lane group kh (0/1)
__device__ __forceinline__ int a_k(int e, int kh) {
    return 16 * kh + e;
}

// ---------------------------------------------------------------- kernel 1a
// Convert weights f32 -> f16, stored TRANSPOSED [n][k] so GEMM B-tiles are
// contiguous-in-k and can be staged with async 16B DMA copies.
__global__ void convert_weightsT(const float* __restrict__ Wq,
                                 const float* __restrict__ Wk,
                                 const float* __restrict__ Wv,
                                 const float* __restrict__ Wo,
                                 _Float16* __restrict__ wqkvT,
                                 _Float16* __restrict__ woT) {
    int i = blockIdx.x * blockDim.x + threadIdx.x;
    if (i < DM * NQKV) {          // i = n*DM + k  (transposed layout)
        int n = i / DM, k = i - n * DM;
        int g = n / DM, c = n - g * DM;
        const float* W = (g == 0) ? Wq : (g == 1) ? Wk : Wv;
        wqkvT[i] = (_Float16)W[k * DM + c];
    }
    if (i < DM * DM) {            // i = n*DM + k
        int n = i / DM, k = i - n * DM;
        woT[i] = (_Float16)Wo[k * DM + n];
    }
}

// ---------------------------------------------------------------- kernel 1b
// Convert activations f32 -> f16 once (f16 copies fit in the 192MB L2,
// so the 8x re-reads in qkv_gemm become L2 hits).
__global__ __launch_bounds__(256) void convert_acts(
    const float* __restrict__ q, const float* __restrict__ k,
    const float* __restrict__ v,
    _Float16* __restrict__ q16, _Float16* __restrict__ k16,
    _Float16* __restrict__ v16) {
    size_t i = ((size_t)blockIdx.x * blockDim.x + threadIdx.x) * 4;
    if (i >= (size_t)MOUT * DM) return;
    const float* s = (blockIdx.y == 0) ? q : (blockIdx.y == 1) ? k : v;
    _Float16*    d = (blockIdx.y == 0) ? q16 : (blockIdx.y == 1) ? k16 : v16;
    float4 x = *(const float4*)(s + i);
    v4h y = {(_Float16)x.x, (_Float16)x.y, (_Float16)x.z, (_Float16)x.w};
    *(v4h*)(d + i) = y;
}

// ---------------------------------------------------------------- kernel 2
// QKV projection GEMM: M = B*KP (edge-padded rows), N = 1536, K = 512
__global__ __launch_bounds__(256) void qkv_gemm(
    const _Float16* __restrict__ q16, const _Float16* __restrict__ k16,
    const _Float16* __restrict__ v16, const _Float16* __restrict__ wqkvT,
    const float* __restrict__ bq, const float* __restrict__ bk,
    const float* __restrict__ bv,
    _Float16* __restrict__ Qh, _Float16* __restrict__ Kh,
    _Float16* __restrict__ Vh) {
    __shared__ _Float16 As[128][36];   // [m][k]
    __shared__ _Float16 Bt[64][40];    // [n][k], 16B-aligned rows

    const int tid  = threadIdx.x;
    const int lane = tid & 31;
    const int wid  = tid >> 5;          // 8 waves -> 8 * 16 = 128 rows
    const int m0    = blockIdx.x * 128;
    const int nbase = blockIdx.y * 64;
    const int group = nbase / DM;       // 0=Q 1=K 2=V
    const _Float16* act = (group == 0) ? q16 : (group == 1) ? k16 : v16;
    const float* bias   = (group == 0) ? bq  : (group == 1) ? bk  : bv;

    v8f acc[4] = {};

    for (int kt = 0; kt < DM / 32; ++kt) {
#ifdef HAVE_ASYNC_LDS
        // stage A: pure f16 copy with edge-pad row clamp
        for (int idx = tid; idx < 128 * 4; idx += 256) {
            int ar = idx >> 2, c = idx & 3;
            int gm = m0 + ar;
            gm = gm < MTOT ? gm : MTOT - 1;     // tail rows are discarded
            int bb  = gm / KP_;
            int p   = gm - bb * KP_;
            int src = p - PAD_;
            src = src < 0 ? 0 : (src > K_ - 1 ? K_ - 1 : src);
            async_copy16(act + ((size_t)(bb * K_ + src)) * DM + kt * 32 + c * 8,
                         &As[ar][c * 8]);
        }
        // stage B: contiguous rows of transposed weights
        for (int idx = tid; idx < 64 * 4; idx += 256) {
            int bn = idx >> 2, c = idx & 3;
            async_copy16(wqkvT + (size_t)(nbase + bn) * DM + kt * 32 + c * 8,
                         &Bt[bn][c * 8]);
        }
        wait_async0();
#else
        for (int idx = tid; idx < 128 * 8; idx += 256) {
            int ar = idx >> 3, ak = (idx & 7) * 4;
            int gm = m0 + ar;
            gm = gm < MTOT ? gm : MTOT - 1;
            int bb  = gm / KP_;
            int p   = gm - bb * KP_;
            int src = p - PAD_;
            src = src < 0 ? 0 : (src > K_ - 1 ? K_ - 1 : src);
            *(v4h*)&As[ar][ak] =
                *(const v4h*)&act[((size_t)(bb * K_ + src)) * DM + kt * 32 + ak];
        }
        for (int idx = tid; idx < 64 * 8; idx += 256) {
            int bn = idx >> 3, c = (idx & 7) * 4;
            *(v4h*)&Bt[bn][c] =
                *(const v4h*)&wqkvT[(size_t)(nbase + bn) * DM + kt * 32 + c];
        }
#endif
        __syncthreads();

        const int kh = lane >> 4;
        const int mr = wid * 16 + (lane & 15);
        v16h a;
#pragma unroll
        for (int e = 0; e < 16; ++e) a[e] = As[mr][a_k(e, kh)];
        v16h bf[4];
#pragma unroll
        for (int sub = 0; sub < 4; ++sub) {
            int n = sub * 16 + (lane & 15);
#pragma unroll
            for (int e = 0; e < 16; ++e) bf[sub][e] = Bt[n][kh * 16 + e];
        }
#pragma unroll
        for (int sub = 0; sub < 4; ++sub)
            acc[sub] = __builtin_amdgcn_wmma_f32_16x16x32_f16(
                false, a, false, bf[sub], (short)0, acc[sub], false, false);
        __syncthreads();
    }

    const int kh = lane >> 4;
    _Float16* dst = (group == 0) ? Qh : (group == 1) ? Kh : Vh;
#pragma unroll
    for (int sub = 0; sub < 4; ++sub) {
        int nc = (nbase - group * DM) + sub * 16 + (lane & 15); // col in group
        float bs = bias[nc];
        int h = nc >> 6, d = nc & 63;
#pragma unroll
        for (int r = 0; r < 8; ++r) {
            int gm = m0 + wid * 16 + kh * 8 + r;
            if (gm < MTOT) {
                int bb = gm / KP_;
                int p  = gm - bb * KP_;
                dst[((size_t)(h * B_ + bb) * KP_ + p) * 64 + d] =
                    (_Float16)(acc[sub][r] + bs);
            }
        }
    }
}

// ---------------------------------------------------------------- kernel 3
// One block per (window, head-batch). 4 waves, 128 threads.
__global__ __launch_bounds__(128) void win_attn(
    const _Float16* __restrict__ Qh, const _Float16* __restrict__ Kh,
    const _Float16* __restrict__ Vh, _Float16* __restrict__ attn) {
    __shared__ _Float16 Kw[WIN_][64];   // [token][dim]  21504 B (contig in k)
    __shared__ _Float16 Vt[64][192];    // [dim][token]  24576 B (contig in k)
    __shared__ _Float16 Qb[16][64];     //               2048 B
    __shared__ float    Srow[16][148];  // f32 scores    9472 B (max col 143)
    __shared__ _Float16 Sh[16][192];    // f16 probs     6144 B
    // total 63744 B

    const int w    = blockIdx.x;
    const int hb   = blockIdx.y;
    const int tid  = threadIdx.x;
    const int lane = tid & 31;
    const int wv   = tid >> 5;          // 0..3
    const size_t base = ((size_t)hb * KP_ + w * STEP_) * 64;

    {   // stage K window (row-major copy: async DMA path if available)
#ifdef HAVE_ASYNC_LDS
        for (int i = tid; i < WIN_ * 64 / 8; i += 128)
            async_copy16(Kh + base + (size_t)i * 8, &Kw[0][0] + i * 8);
#else
        const v4h* src = (const v4h*)(Kh + base);
        v4h* dk = (v4h*)&Kw[0][0];
        for (int i = tid; i < WIN_ * 64 / 4; i += 128) dk[i] = src[i];
#endif
        // stage V window transposed: Vt[dim][token] (element scatter)
        for (int i = tid; i < WIN_ * 64 / 4; i += 128) {
            int t = (i * 4) >> 6;            // token
            int d0 = (i * 4) & 63;           // dim
            v4h vv = ((const v4h*)(Vh + base))[i];
            Vt[d0 + 0][t] = vv[0];
            Vt[d0 + 1][t] = vv[1];
            Vt[d0 + 2][t] = vv[2];
            Vt[d0 + 3][t] = vv[3];
        }
        // zero the K-dim pad (tokens 168..191)
        for (int i = tid; i < 64 * 24; i += 128) {
            Vt[i / 24][WIN_ + (i % 24)] = (_Float16)0.f;
        }
        if (w + 1 < NWIN)   // warm L2 for the next window's K slab
            __builtin_prefetch(Kh + base + (size_t)STEP_ * 64, 0, 1);
#ifdef HAVE_ASYNC_LDS
        wait_async0();
#endif
    }
    __syncthreads();

    const float scale = 0.07715167498104595f;   // 1/sqrt(168)

    for (int rb = 0; rb < 6; ++rb) {
        const int qr0 = PAD_ + rb * 16;         // 42..122 (rows 42..137)
        {   // stage Q block
#ifdef HAVE_ASYNC_LDS
            if (tid < 16 * 64 / 8)
                async_copy16(Qh + base + (size_t)qr0 * 64 + (size_t)tid * 8,
                             &Qb[0][0] + tid * 8);
            wait_async0();
#else
            const v4h* srcq = (const v4h*)(Qh + base + (size_t)qr0 * 64);
            v4h* dq = (v4h*)&Qb[0][0];
            for (int i = tid; i < 16 * 64 / 4; i += 128) dq[i] = srcq[i];
#endif
        }
        __syncthreads();

        // ---- S = Qb @ Kw^T (skip fully-masked col tiles)
        const int kh  = lane >> 4;
        const int nct = ((qr0 + 15) >> 4) + 1;  // tiles needed (max 9 -> col 143)
        v16h aq[2];                             // hoisted: Q frags for ks=0,1
#pragma unroll
        for (int ks = 0; ks < 2; ++ks)
#pragma unroll
            for (int e = 0; e < 16; ++e)
                aq[ks][e] = Qb[lane & 15][ks * 32 + a_k(e, kh)];
        for (int ct = wv; ct < nct; ct += 4) {
            v8f acc = {};
            int n = ct * 16 + (lane & 15);
            int ncl = n < WIN_ ? n : WIN_ - 1;
#pragma unroll
            for (int ks = 0; ks < 2; ++ks) {
                v16h bf;
#pragma unroll
                for (int e = 0; e < 16; ++e)
                    bf[e] = Kw[ncl][ks * 32 + kh * 16 + e];
                acc = __builtin_amdgcn_wmma_f32_16x16x32_f16(
                    false, aq[ks], false, bf, (short)0, acc, false, false);
            }
#pragma unroll
            for (int r = 0; r < 8; ++r) Srow[kh * 8 + r][n] = acc[r] * scale;
        }
        __syncthreads();

        // ---- causal softmax: 8 lanes per row, shfl_xor reductions
        {
            const int row = tid >> 3;           // 0..15
            const int sub = tid & 7;            // 0..7
            const int i   = qr0 + row;          // causal bound (<= 137)
            float mx = -3.0e38f;
            for (int j = sub; j <= i; j += 8) mx = fmaxf(mx, Srow[row][j]);
            mx = fmaxf(mx, __shfl_xor(mx, 1));
            mx = fmaxf(mx, __shfl_xor(mx, 2));
            mx = fmaxf(mx, __shfl_xor(mx, 4));
            float sum = 0.f;
            for (int j = sub; j <= i; j += 8) {
                float e = __expf(Srow[row][j] - mx);
                Srow[row][j] = e;
                sum += e;
            }
            sum += __shfl_xor(sum, 1);
            sum += __shfl_xor(sum, 2);
            sum += __shfl_xor(sum, 4);
            float inv = 1.f / sum;
            for (int j = sub; j < 192; j += 8) {
                float val = (j <= i) ? Srow[row][j] * inv : 0.f;
                Sh[row][j] = (_Float16)val;
            }
        }
        __syncthreads();

        // ---- O = P @ Vt^T : 16 x 64, one 16-col tile per wave, K = 192
        {
            v8f acc = {};
            int n = wv * 16 + (lane & 15);
#pragma unroll
            for (int ks = 0; ks < 6; ++ks) {
                v16h a, bf;
#pragma unroll
                for (int e = 0; e < 16; ++e)
                    a[e] = Sh[lane & 15][ks * 32 + a_k(e, kh)];
#pragma unroll
                for (int e = 0; e < 16; ++e)
                    bf[e] = Vt[n][ks * 32 + kh * 16 + e];
                acc = __builtin_amdgcn_wmma_f32_16x16x32_f16(
                    false, a, false, bf, (short)0, acc, false, false);
            }
            int h = hb >> 2, bb = hb & 3;
            int d = n;
#pragma unroll
            for (int r = 0; r < 8; ++r) {
                int lrow = rb * 16 + kh * 8 + r;       // 0..95
                if (lrow < STEP_) {
                    int t = w * STEP_ + lrow;
                    attn[((size_t)(bb * K_ + t)) * 512 + h * 64 + d] =
                        (_Float16)acc[r];
                }
            }
        }
        __syncthreads();
    }
}

// ---------------------------------------------------------------- kernel 4
// Output projection: M = B*K, N = 512, K = 512, f32 out + bias
__global__ __launch_bounds__(256) void out_gemm(
    const _Float16* __restrict__ attn, const _Float16* __restrict__ woT,
    const float* __restrict__ bo, float* __restrict__ out) {
    __shared__ _Float16 As[128][36];
    __shared__ _Float16 Bt[64][40];

    const int tid  = threadIdx.x;
    const int lane = tid & 31;
    const int wid  = tid >> 5;
    const int m0    = blockIdx.x * 128;
    const int nbase = blockIdx.y * 64;

    v8f acc[4] = {};

    for (int kt = 0; kt < DM / 32; ++kt) {
#ifdef HAVE_ASYNC_LDS
        for (int idx = tid; idx < 128 * 4; idx += 256) {
            int ar = idx >> 2, c = idx & 3;
            int gm = m0 + ar;
            gm = gm < MOUT ? gm : MOUT - 1;     // clamped rows are discarded
            async_copy16(attn + (size_t)gm * 512 + kt * 32 + c * 8,
                         &As[ar][c * 8]);
        }
        for (int idx = tid; idx < 64 * 4; idx += 256) {
            int bn = idx >> 2, c = idx & 3;
            async_copy16(woT + (size_t)(nbase + bn) * DM + kt * 32 + c * 8,
                         &Bt[bn][c * 8]);
        }
        wait_async0();
#else
        for (int idx = tid; idx < 128 * 8; idx += 256) {
            int ar = idx >> 3, ak = (idx & 7) * 4;
            int gm = m0 + ar;
            gm = gm < MOUT ? gm : MOUT - 1;
            *(v4h*)&As[ar][ak] =
                *(const v4h*)&attn[(size_t)gm * 512 + kt * 32 + ak];
        }
        for (int idx = tid; idx < 64 * 8; idx += 256) {
            int bn = idx >> 3, c = (idx & 7) * 4;
            *(v4h*)&Bt[bn][c] =
                *(const v4h*)&woT[(size_t)(nbase + bn) * DM + kt * 32 + c];
        }
#endif
        __syncthreads();

        const int kh = lane >> 4;
        const int mr = wid * 16 + (lane & 15);
        v16h a;
#pragma unroll
        for (int e = 0; e < 16; ++e) a[e] = As[mr][a_k(e, kh)];
        v16h bf[4];
#pragma unroll
        for (int sub = 0; sub < 4; ++sub) {
            int n = sub * 16 + (lane & 15);
#pragma unroll
            for (int e = 0; e < 16; ++e) bf[sub][e] = Bt[n][kh * 16 + e];
        }
#pragma unroll
        for (int sub = 0; sub < 4; ++sub)
            acc[sub] = __builtin_amdgcn_wmma_f32_16x16x32_f16(
                false, a, false, bf[sub], (short)0, acc[sub], false, false);
        __syncthreads();
    }

    const int kh = lane >> 4;
#pragma unroll
    for (int sub = 0; sub < 4; ++sub) {
        int n = nbase + sub * 16 + (lane & 15);
        float bs = bo[n];
#pragma unroll
        for (int r = 0; r < 8; ++r) {
            int gm = m0 + wid * 16 + kh * 8 + r;
            if (gm < MOUT) out[(size_t)gm * 512 + n] = acc[sub][r] + bs;
        }
    }
}

// ---------------------------------------------------------------- launch
extern "C" void kernel_launch(void* const* d_in, const int* in_sizes, int n_in,
                              void* d_out, int out_size, void* d_ws,
                              size_t ws_size, hipStream_t stream) {
    const float* query = (const float*)d_in[0];
    const float* key   = (const float*)d_in[1];
    const float* value = (const float*)d_in[2];
    const float* W_q   = (const float*)d_in[3];
    const float* b_q   = (const float*)d_in[4];
    const float* W_k   = (const float*)d_in[5];
    const float* b_k   = (const float*)d_in[6];
    const float* W_v   = (const float*)d_in[7];
    const float* b_v   = (const float*)d_in[8];
    const float* W_o   = (const float*)d_in[9];
    const float* b_o   = (const float*)d_in[10];

    _Float16* ws    = (_Float16*)d_ws;
    _Float16* wqkvT = ws;                                   // 1536*512
    _Float16* woT   = wqkvT + (size_t)DM * NQKV;            // 512*512
    _Float16* Qh    = woT + (size_t)DM * DM;                // 32*8484*64
    _Float16* Kh    = Qh + (size_t)HB_ * KP_ * 64;
    _Float16* Vh    = Kh + (size_t)HB_ * KP_ * 64;
    _Float16* attn  = Vh + (size_t)HB_ * KP_ * 64;          // 4*8400*512
    _Float16* q16   = attn + (size_t)MOUT * DM;             // 4*8400*512
    _Float16* k16   = q16 + (size_t)MOUT * DM;
    _Float16* v16   = k16 + (size_t)MOUT * DM;

    convert_weightsT<<<(DM * NQKV + 255) / 256, 256, 0, stream>>>(
        W_q, W_k, W_v, W_o, wqkvT, woT);

    dim3 gc(((size_t)MOUT * DM / 4 + 255) / 256, 3);
    convert_acts<<<gc, 256, 0, stream>>>(query, key, value, q16, k16, v16);

    dim3 g2((MTOT + 127) / 128, NQKV / 64);
    qkv_gemm<<<g2, 256, 0, stream>>>(q16, k16, v16, wqkvT,
                                     b_q, b_k, b_v, Qh, Kh, Vh);

    dim3 g3(NWIN, HB_);
    win_attn<<<g3, 128, 0, stream>>>(Qh, Kh, Vh, attn);

    dim3 g4((MOUT + 127) / 128, DM / 64);
    out_gemm<<<g4, 256, 0, stream>>>(attn, woT, b_o, (float*)d_out);
}
